// TemporalGNN_41558103556325
// MI455X (gfx1250) — matrix operand order repaired and checked by reading, not compiled, not run on hardware
//
#include <hip/hip_runtime.h>
#include <hip/hip_bf16.h>

typedef __attribute__((ext_vector_type(16))) __bf16 v16bf;
typedef __attribute__((ext_vector_type(8)))  float  v8f;

// hardware f32 -> bf16 convert (v_cvt_*bf16_f32), RNE
__device__ __forceinline__ __bf16 f2bf(float f) { return (__bf16)f; }

// ---------------------------------------------------------------------------
// Graph aggregation kernels (memory-bound path)
// ---------------------------------------------------------------------------
__global__ void deg_kernel(const int* __restrict__ dst, float* __restrict__ deg, int E) {
    int e = blockIdx.x * blockDim.x + threadIdx.x;
    if (e < E) atomicAdd(&deg[dst[e]], 1.0f);
}

// one edge handled by 32 threads, 4 features each (D = 128)
__global__ void scatter_add_kernel(const float* __restrict__ X,
                                   const int* __restrict__ src,
                                   const int* __restrict__ dst,
                                   float* __restrict__ agg, int E) {
    int t = blockIdx.x * blockDim.x + threadIdx.x;
    int e = t >> 5;
    if (e >= E) return;
    int c = (t & 31) << 2;
    const float4 xs = *(const float4*)(X + (size_t)src[e] * 128 + c);
    float*       ad = agg + (size_t)dst[e] * 128 + c;
    atomicAdd(&ad[0], xs.x);
    atomicAdd(&ad[1], xs.y);
    atomicAdd(&ad[2], xs.z);
    atomicAdd(&ad[3], xs.w);
}

__global__ void mean_kernel(float* __restrict__ agg, const float* __restrict__ deg, int total) {
    int i = blockIdx.x * blockDim.x + threadIdx.x;
    if (i < total) agg[i] = agg[i] / fmaxf(deg[i >> 7], 1.0f);
}

// ---------------------------------------------------------------------------
// Fused WMMA GEMM:  Out = act( A1 @ W1^T  [+ A2 @ W2^T]  + bias1 [+ bias2] )
// A: [M,K] f32 row-major, W: [Nout,K] f32 row-major (PyTorch linear layout).
// Each wave computes one 16x16 output tile; K consumed 32 at a time via
// v_wmma_f32_16x16x32_bf16 (bf16 inputs, f32 accumulate).
// ---------------------------------------------------------------------------
__device__ __forceinline__ v16bf load_a_frag(const float* __restrict__ A, int ldA,
                                             int row0, int kc, int lane) {
    // 16-bit A 16x32 layout: lanes 0-15 -> M=0..15 / K-half 0, lanes 16-31 -> K-half 8.
    // Per lane this is two contiguous 8-float runs: k = base+[0..7] and base+[16..23].
    int m     = row0 + (lane & 15);
    int khalf = (lane >> 4) << 3;            // 0 or 8
    const float* base = A + (size_t)m * ldA + kc + khalf;
    float4 lo0 = *(const float4*)(base + 0);
    float4 lo1 = *(const float4*)(base + 4);
    float4 hi0 = *(const float4*)(base + 16);
    float4 hi1 = *(const float4*)(base + 20);
    v16bf f;
    f[0]  = f2bf(lo0.x); f[1]  = f2bf(lo0.y); f[2]  = f2bf(lo0.z); f[3]  = f2bf(lo0.w);
    f[4]  = f2bf(lo1.x); f[5]  = f2bf(lo1.y); f[6]  = f2bf(lo1.z); f[7]  = f2bf(lo1.w);
    f[8]  = f2bf(hi0.x); f[9]  = f2bf(hi0.y); f[10] = f2bf(hi0.z); f[11] = f2bf(hi0.w);
    f[12] = f2bf(hi1.x); f[13] = f2bf(hi1.y); f[14] = f2bf(hi1.z); f[15] = f2bf(hi1.w);
    return f;
}

__device__ __forceinline__ v16bf load_b_frag(const float* __restrict__ W, int ldW,
                                             int col0, int kc, int lane) {
    // B 32x16 (KxN): lanes 0-15 -> K=0..7,16..23, lanes 16-31 -> K=8..15,24..31; col = lane&15.
    // Per lane: two contiguous 8-float runs of W row: k = base+[0..7] and base+[16..23].
    int n     = col0 + (lane & 15);
    int kbase = kc + ((lane >> 4) << 3);     // 0 or 8
    const float* base = W + (size_t)n * ldW + kbase;
    float4 q0 = *(const float4*)(base + 0);
    float4 q1 = *(const float4*)(base + 4);
    float4 q2 = *(const float4*)(base + 16);
    float4 q3 = *(const float4*)(base + 20);
    v16bf f;
    f[0]  = f2bf(q0.x); f[1]  = f2bf(q0.y); f[2]  = f2bf(q0.z); f[3]  = f2bf(q0.w);
    f[4]  = f2bf(q1.x); f[5]  = f2bf(q1.y); f[6]  = f2bf(q1.z); f[7]  = f2bf(q1.w);
    f[8]  = f2bf(q2.x); f[9]  = f2bf(q2.y); f[10] = f2bf(q2.z); f[11] = f2bf(q2.w);
    f[12] = f2bf(q3.x); f[13] = f2bf(q3.y); f[14] = f2bf(q3.z); f[15] = f2bf(q3.w);
    return f;
}

__global__ void gemm_wmma_kernel(const float* __restrict__ A1, const float* __restrict__ W1,
                                 const float* __restrict__ A2, const float* __restrict__ W2,
                                 const float* __restrict__ bias1, const float* __restrict__ bias2,
                                 float* __restrict__ Out,
                                 int M, int K, int Nout, int do_relu) {
    int flat   = blockIdx.x * blockDim.x + threadIdx.x;
    int wave   = flat >> 5;                  // wave-uniform tile id (EXEC all-ones)
    int lane   = threadIdx.x & 31;
    int tilesN = Nout >> 4;
    int tilesM = M >> 4;
    if (wave >= tilesM * tilesN) return;     // uniform across the wave
    int tm = wave / tilesN, tn = wave % tilesN;
    int row0 = tm << 4, col0 = tn << 4;

    v8f acc = {};
#pragma unroll 4
    for (int kc = 0; kc < K; kc += 32) {
        v16bf a = load_a_frag(A1, K, row0, kc, lane);
        v16bf b = load_b_frag(W1, K, col0, kc, lane);
        acc = __builtin_amdgcn_wmma_f32_16x16x32_bf16(false, a, false, b,
                                                      (short)0, acc, false, false);
    }
    if (A2) {
#pragma unroll 4
        for (int kc = 0; kc < K; kc += 32) {
            v16bf a = load_a_frag(A2, K, row0, kc, lane);
            v16bf b = load_b_frag(W2, K, col0, kc, lane);
            acc = __builtin_amdgcn_wmma_f32_16x16x32_bf16(false, a, false, b,
                                                          (short)0, acc, false, false);
        }
    }

    int   n    = col0 + (lane & 15);
    float bcol = (bias1 ? bias1[n] : 0.0f) + (bias2 ? bias2[n] : 0.0f);
    int   mb   = (lane >> 4) << 3;           // f32 C/D layout: VGPR r -> M=r (+8 for hi lanes)
#pragma unroll
    for (int r = 0; r < 8; ++r) {
        float v = acc[r] + bcol;
        if (do_relu) v = fmaxf(v, 0.0f);
        Out[(size_t)(row0 + mb + r) * Nout + n] = v;
    }
}

// ---------------------------------------------------------------------------
// Sequential LSTM over N steps. One workgroup, w_hh resident in the 320KB WGP
// LDS (512*128 f32 = 256KB). 512 threads: thread g computes gate pre-act g.
// Output projection (H=128 -> C=2) fused via ds_add_f32 LDS atomics.
// gates_pre[t, g] already holds x_t @ w_ih^T + b_ih + b_hh.
// ---------------------------------------------------------------------------
__global__ void lstm_seq_kernel(const float* __restrict__ gates_pre,
                                const float* __restrict__ w_hh,
                                const float* __restrict__ w_out,
                                const float* __restrict__ b_out,
                                float* __restrict__ out, int N) {
    extern __shared__ float smem[];
    float* whh    = smem;                 // 512*128
    float* h      = whh + 512 * 128;      // 128
    float* c      = h + 128;              // 128
    float* pre    = c + 128;              // 512
    float* outred = pre + 512;            // 2
    const int tid = threadIdx.x;          // blockDim.x == 512

    for (int i = tid; i < 512 * 128; i += blockDim.x) whh[i] = w_hh[i];
    if (tid < 128) { h[tid] = 0.0f; c[tid] = 0.0f; }
    if (tid < 2) outred[tid] = 0.0f;
    __syncthreads();

    const float* wrow = &whh[tid * 128];
    for (int t = 0; t < N; ++t) {
        float acc = gates_pre[(size_t)t * 512 + tid];
#pragma unroll 8
        for (int k = 0; k < 128; ++k) acc = fmaf(h[k], wrow[k], acc);
        pre[tid] = acc;
        __syncthreads();

        if (tid < 128) {
            float i_ = 1.0f / (1.0f + __expf(-pre[tid]));
            float f_ = 1.0f / (1.0f + __expf(-pre[tid + 128]));
            float g_ = tanhf(pre[tid + 256]);
            float o_ = 1.0f / (1.0f + __expf(-pre[tid + 384]));
            float cc = f_ * c[tid] + i_ * g_;
            c[tid] = cc;
            float hh = o_ * tanhf(cc);
            h[tid] = hh;
            atomicAdd(&outred[0], hh * w_out[tid]);        // ds_add_f32
            atomicAdd(&outred[1], hh * w_out[128 + tid]);
        }
        __syncthreads();
        if (tid < 2) {
            out[(size_t)t * 2 + tid] = outred[tid] + b_out[tid];
            outred[tid] = 0.0f;
        }
        __syncthreads();
    }
}

// ---------------------------------------------------------------------------
extern "C" void kernel_launch(void* const* d_in, const int* in_sizes, int n_in,
                              void* d_out, int out_size, void* d_ws, size_t ws_size,
                              hipStream_t stream) {
    const float* x     = (const float*)d_in[0];
    const int*   ei    = (const int*)d_in[1];
    const float* w_l0  = (const float*)d_in[2];
    const float* b_l0  = (const float*)d_in[3];
    const float* w_r0  = (const float*)d_in[4];
    const float* w_l1  = (const float*)d_in[5];
    const float* b_l1  = (const float*)d_in[6];
    const float* w_r1  = (const float*)d_in[7];
    const float* w_ih  = (const float*)d_in[8];
    const float* w_hh  = (const float*)d_in[9];
    const float* b_ih  = (const float*)d_in[10];
    const float* b_hh  = (const float*)d_in[11];
    const float* w_out = (const float*)d_in[12];
    const float* b_out = (const float*)d_in[13];
    float* out = (float*)d_out;

    const int N = in_sizes[0] / 128;
    const int E = in_sizes[1] / 2;
    const int*   src = ei;
    const int*   dst = ei + E;

    // workspace layout (floats)
    float* ws    = (float*)d_ws;
    float* deg   = ws;                          // N
    float* agg   = deg + N;                     // N*128
    float* h0    = agg + (size_t)N * 128;       // N*128
    float* h1    = h0  + (size_t)N * 128;       // N*128
    float* gates = h1  + (size_t)N * 128;       // N*512

    const int THREADS = 256;
    auto cdiv = [](long long a, long long b) { return (int)((a + b - 1) / b); };

    // ---- degree (shared by both layers) ----
    hipMemsetAsync(deg, 0, (size_t)N * sizeof(float), stream);
    deg_kernel<<<cdiv(E, THREADS), THREADS, 0, stream>>>(dst, deg, E);

    // ---- SAGE layer 0 ----
    hipMemsetAsync(agg, 0, (size_t)N * 128 * sizeof(float), stream);
    scatter_add_kernel<<<cdiv((long long)E * 32, THREADS), THREADS, 0, stream>>>(x, src, dst, agg, E);
    mean_kernel<<<cdiv((long long)N * 128, THREADS), THREADS, 0, stream>>>(agg, deg, N * 128);
    {
        int tiles = (N >> 4) * (128 >> 4);
        gemm_wmma_kernel<<<cdiv(tiles, THREADS / 32), THREADS, 0, stream>>>(
            agg, w_l0, x, w_r0, b_l0, nullptr, h0, N, 128, 128, /*relu=*/1);
    }

    // ---- SAGE layer 1 ----
    hipMemsetAsync(agg, 0, (size_t)N * 128 * sizeof(float), stream);
    scatter_add_kernel<<<cdiv((long long)E * 32, THREADS), THREADS, 0, stream>>>(h0, src, dst, agg, E);
    mean_kernel<<<cdiv((long long)N * 128, THREADS), THREADS, 0, stream>>>(agg, deg, N * 128);
    {
        int tiles = (N >> 4) * (128 >> 4);
        gemm_wmma_kernel<<<cdiv(tiles, THREADS / 32), THREADS, 0, stream>>>(
            agg, w_l1, h0, w_r1, b_l1, nullptr, h1, N, 128, 128, /*relu=*/1);
    }

    // ---- LSTM input projection: gates = h1 @ w_ih^T + b_ih + b_hh ----
    {
        int tiles = (N >> 4) * (512 >> 4);
        gemm_wmma_kernel<<<cdiv(tiles, THREADS / 32), THREADS, 0, stream>>>(
            h1, w_ih, nullptr, nullptr, b_ih, b_hh, gates, N, 128, 512, /*relu=*/0);
    }

    // ---- sequential LSTM + fused output projection ----
    {
        size_t smem = (size_t)(512 * 128 + 128 + 128 + 512 + 2) * sizeof(float); // ~259 KB < 320 KB WGP LDS
        lstm_seq_kernel<<<1, 512, smem, stream>>>(gates, w_hh, w_out, b_out, out, N);
    }
}
